// KGAtt_49082886259211
// MI455X (gfx1250) — matrix operand order repaired and checked by reading, not compiled, not run on hardware
//
#include <hip/hip_runtime.h>
#include <hip/hip_bf16.h>

#define N_ENT   100000
#define N_RELS  237
#define IN_DIM  128
#define E_EDGES 400000
#define NTILES  (E_EDGES / 16)      // 25000, exact
#define HENT    (N_ENT * IN_DIM)    // 12,800,000
#define HREL    (N_RELS * IN_DIM)   // 30,336
#define OUT_TOTAL (HENT + HREL)
#define BPACK_ELEMS (12 * 8 * 32 * 16)  // 49152 bf16 = 96KB

typedef __attribute__((ext_vector_type(16))) __bf16 v16bf;
typedef __attribute__((ext_vector_type(16))) float  v16f;
typedef __attribute__((ext_vector_type(8)))  float  v8f;

__device__ __forceinline__ unsigned short f2bf_bits(float f) {
    unsigned u = __float_as_uint(f);
    unsigned r = u + 0x7FFFu + ((u >> 16) & 1u);   // round-to-nearest-even
    return (unsigned short)(r >> 16);
}

// ---------------------------------------------------------------- zero init
__global__ void kg_zero(float* __restrict__ out, float* __restrict__ ebsum,
                        float* __restrict__ relcnt) {
    int i = blockIdx.x * blockDim.x + threadIdx.x;
    int stride = gridDim.x * blockDim.x;
    for (int j = i; j < OUT_TOTAL; j += stride) out[j] = 0.0f;
    for (int j = i; j < N_ENT;    j += stride) ebsum[j] = 0.0f;
    for (int j = i; j < N_RELS;   j += stride) relcnt[j] = 0.0f;
}

// ---------------------------------------------------- pack Wa -> bf16 B-frags
// B tile (kb,nt) is 32x16 (KxN): lane%16 = N, element j = K (table order: 8*(lane/16)+j, +16 for j>=8).
// B[k][n] = Wa[n][k]  (c = h @ Wa^T)
__global__ void kg_pack(const float* __restrict__ Wa, unsigned short* __restrict__ Bp) {
    int idx = blockIdx.x * blockDim.x + threadIdx.x;
    if (idx >= BPACK_ELEMS) return;
    int j    = idx & 15;
    int lane = (idx >> 4) & 31;
    int tI   = idx >> 9;          // kb*8 + nt
    int nt   = tI & 7;
    int kb   = tI >> 3;
    int k = kb * 32 + ((j >> 3) << 4) + ((lane >> 4) << 3) + (j & 7);
    int n = nt * 16 + (lane & 15);
    Bp[idx] = f2bf_bits(Wa[n * 384 + k]);
}

// 4 k-blocks over one 128-float segment with fixed base pointer (no selection)
__device__ __forceinline__ void kg_seg(const float* __restrict__ p, int kbBase,
                                       int half, int lane,
                                       const unsigned short* __restrict__ Bp,
                                       v8f (&acc)[8]) {
#pragma unroll 1
    for (int k4 = 0; k4 < 4; ++k4) {
        const int off = k4 * 32 + half * 8;     // K = off+{0..7}, off+{16..23}
        float4 f0 = ((const float4*)(p + off))[0];
        float4 f1 = ((const float4*)(p + off))[1];
        float4 f2 = ((const float4*)(p + off + 16))[0];
        float4 f3 = ((const float4*)(p + off + 16))[1];
        v16f af;
        af[0]  = f0.x; af[1]  = f0.y; af[2]  = f0.z; af[3]  = f0.w;
        af[4]  = f1.x; af[5]  = f1.y; af[6]  = f1.z; af[7]  = f1.w;
        af[8]  = f2.x; af[9]  = f2.y; af[10] = f2.z; af[11] = f2.w;
        af[12] = f3.x; af[13] = f3.y; af[14] = f3.z; af[15] = f3.w;
        v16bf a = __builtin_convertvector(af, v16bf);   // v_cvt_pk_bf16_f32

        const int kb = kbBase + k4;
#pragma unroll
        for (int nt = 0; nt < 8; ++nt) {
            v16bf b = *(const v16bf*)(Bp + (((kb * 8 + nt) * 32 + lane) << 4));
            acc[nt] = __builtin_amdgcn_wmma_f32_16x16x32_bf16(
                false, a, false, b, (short)0, acc[nt], false, false);
        }
    }
}

// ------------------------------------------------------------- fused main
__global__ __launch_bounds__(256) void kg_main(
    const int*   __restrict__ trip,   const float* __restrict__ ent,
    const float* __restrict__ rele,   const float* __restrict__ ba,
    const float* __restrict__ Wa2,    const float* __restrict__ ba2,
    const unsigned short* __restrict__ Bp,
    float* __restrict__ out, float* __restrict__ ebsum, float* __restrict__ relcnt)
{
    extern __shared__ float smem[];          // rel accumulators: 237*128 + 237 floats
    float* ldsRel = smem;
    float* ldsCnt = smem + N_RELS * IN_DIM;

    for (int i = threadIdx.x; i < N_RELS * IN_DIM + N_RELS; i += blockDim.x) smem[i] = 0.0f;
    __syncthreads();

    const int lane = threadIdx.x & 31;
    const int wid  = threadIdx.x >> 5;
    const int gw   = blockIdx.x * (blockDim.x >> 5) + wid;
    const int GW   = gridDim.x * (blockDim.x >> 5);
    const int r    = lane & 15;      // row within tile / N within n-tile
    const int half = lane >> 4;

    // hoisted epilogue constants (per-lane column r)
    float ba_f[8], wa2_f[8];
#pragma unroll
    for (int nt = 0; nt < 8; ++nt) {
        ba_f[nt]  = ba[nt * 16 + r];
        wa2_f[nt] = Wa2[nt * 16 + r];
    }
    const float ba2v = ba2[0];

    int t = gw;
    int srcv = 0, dstv = 0, relv = 0;
    if (t < NTILES) {
        int e = t * 16 + r;
        srcv = trip[e * 3 + 0];
        dstv = trip[e * 3 + 1];
        relv = trip[e * 3 + 2];
    }

    for (; t < NTILES; t += GW) {
        // ---- software pipeline: fetch next tile's triplets early + prefetch rows
        const int tn = t + GW;
        int srcn = srcv, dstn = dstv, reln = relv;
        if (tn < NTILES) {
            int en = tn * 16 + r;
            srcn = trip[en * 3 + 0];
            dstn = trip[en * 3 + 1];
            reln = trip[en * 3 + 2];
            const char* ps = (const char*)(ent  + (size_t)srcn * IN_DIM);
            const char* pd = (const char*)(ent  + (size_t)dstn * IN_DIM);
            const char* pr = (const char*)(rele + (size_t)reln * IN_DIM);
#pragma unroll
            for (int o = 0; o < 512; o += 128) {     // 512B row, 128B lines
                __builtin_prefetch(ps + o, 0, 3);    // global_prefetch_b8
                __builtin_prefetch(pd + o, 0, 3);
                __builtin_prefetch(pr + o, 0, 3);
            }
        }

        const float* p0 = ent  + (size_t)srcv * IN_DIM;
        const float* p1 = rele + (size_t)relv * IN_DIM;
        const float* p2 = ent  + (size_t)dstv * IN_DIM;

        v8f acc[8] = {};

        // K = 384 split by h-concat segment: [ent[src] | rel[rel] | ent[dst]]
        kg_seg(p0, 0, half, lane, Bp, acc);
        kg_seg(p1, 4, half, lane, Bp, acc);
        kg_seg(p2, 8, half, lane, Bp, acc);

        // epilogue: c = acc + ba; b = lrelu(c.Wa2 + ba2); e = exp(b); scatter e*c
#pragma unroll
        for (int g = 0; g < 8; ++g) {
            float tsum = 0.0f;
#pragma unroll
            for (int nt = 0; nt < 8; ++nt) {
                float c = acc[nt][g] + ba_f[nt];
                acc[nt][g] = c;
                tsum += c * wa2_f[nt];
            }
            // reduce across the 16 lanes of this half (row M = g + 8*half)
            tsum += __shfl_xor(tsum, 1, 32);
            tsum += __shfl_xor(tsum, 2, 32);
            tsum += __shfl_xor(tsum, 4, 32);
            tsum += __shfl_xor(tsum, 8, 32);
            float bv = tsum + ba2v;
            bv = (bv > 0.0f) ? bv : 0.01f * bv;        // leaky_relu(0.01)
            float eb = __expf(bv);

            int m    = g + (half << 3);
            int srcm = __shfl(srcv, m, 32);
            int relm = __shfl(relv, m, 32);
            if (r == 0) {
                atomicAdd(ebsum + srcm, eb);
                atomicAdd(ldsCnt + relm, 1.0f);
            }
            float* hrow = out    + (size_t)srcm * IN_DIM + r;
            float* rrow = ldsRel + relm * IN_DIM + r;
#pragma unroll
            for (int nt = 0; nt < 8; ++nt) {
                float v = eb * acc[nt][g];
                atomicAdd(hrow + nt * 16, v);      // global_atomic_add_f32 (L2-resident)
                atomicAdd(rrow + nt * 16, v);      // ds_add_f32
            }
        }

        srcv = srcn; dstv = dstn; relv = reln;
    }

    // flush LDS rel accumulators once per block
    __syncthreads();
    float* outRel = out + HENT;
    for (int i = threadIdx.x; i < N_RELS * IN_DIM; i += blockDim.x)
        atomicAdd(outRel + i, ldsRel[i]);
    for (int i = threadIdx.x; i < N_RELS; i += blockDim.x)
        atomicAdd(relcnt + i, ldsCnt[i]);
}

// ------------------------------------------------------------- finalize
__global__ void kg_final(float* __restrict__ out, const float* __restrict__ ebsum,
                         const float* __restrict__ relcnt) {
    int i = blockIdx.x * blockDim.x + threadIdx.x;
    int stride = gridDim.x * blockDim.x;
    for (int j = i; j < OUT_TOTAL; j += stride) {
        float x;
        if (j < HENT) {
            float s = ebsum[j >> 7];
            s = (s == 0.0f) ? 1e-12f : s;
            x = out[j] / s;
        } else {
            int jr = j - HENT;
            float c = fmaxf(relcnt[jr >> 7], 1.0f);
            x = out[j] / c;
        }
        out[j] = (x > 0.0f) ? x : expm1f(x);   // ELU
    }
}

extern "C" void kernel_launch(void* const* d_in, const int* in_sizes, int n_in,
                              void* d_out, int out_size, void* d_ws, size_t ws_size,
                              hipStream_t stream) {
    const int*   trip = (const int*)d_in[0];
    const float* ent  = (const float*)d_in[1];
    const float* rele = (const float*)d_in[2];
    const float* Wa   = (const float*)d_in[3];
    const float* ba   = (const float*)d_in[4];
    const float* Wa2  = (const float*)d_in[5];
    const float* ba2  = (const float*)d_in[6];
    float* out = (float*)d_out;

    unsigned short* Bp     = (unsigned short*)d_ws;                 // 96 KB
    float*          ebsum  = (float*)((char*)d_ws + BPACK_ELEMS * 2);
    float*          relcnt = ebsum + N_ENT;

    kg_zero<<<4096, 256, 0, stream>>>(out, ebsum, relcnt);
    kg_pack<<<(BPACK_ELEMS + 255) / 256, 256, 0, stream>>>(Wa, Bp);

    size_t shmem = (size_t)(N_RELS * IN_DIM + N_RELS) * sizeof(float);  // ~122 KB of 320 KB LDS
    kg_main<<<256, 256, shmem, stream>>>(trip, ent, rele, ba, Wa2, ba2, Bp,
                                         out, ebsum, relcnt);

    kg_final<<<(OUT_TOTAL + 255) / 256, 256, 0, stream>>>(out, ebsum, relcnt);
}
